// MLPToGNNV4_63548336111664
// MI455X (gfx1250) — matrix operand order repaired and checked by reading, not compiled, not run on hardware
//
#include <hip/hip_runtime.h>

// ---------------------------------------------------------------------------
// Problem constants: B=128, N=64, S=256, A=64, DK=64, DV=32, H=128
// ---------------------------------------------------------------------------
#define CB 128
#define CN 64
#define CS 256
#define CA 64
#define CDK 64
#define CDV 32
#define CH 128

typedef __bf16 bf16_t;
typedef __attribute__((ext_vector_type(16))) __bf16 v16bf;
typedef __attribute__((ext_vector_type(8)))  __bf16 v8bf;
typedef __attribute__((ext_vector_type(8)))  float  v8f;
typedef __attribute__((ext_vector_type(4))) unsigned int u32x4;
typedef __attribute__((ext_vector_type(8))) int i32x8;
typedef __attribute__((ext_vector_type(4))) int i32x4;

#define WAVE (threadIdx.x >> 5)

// ---- CDNA5 async / TDM feature guards (fallback = plain vector copies) ----
#if defined(__HIP_DEVICE_COMPILE__)
#if __has_builtin(__builtin_amdgcn_global_load_async_to_lds_b128) && \
    __has_builtin(__builtin_amdgcn_s_wait_asynccnt)
#define HAVE_ASYNC_LDS 1
#else
#define HAVE_ASYNC_LDS 0
#endif
#if __has_builtin(__builtin_amdgcn_tensor_load_to_lds) && \
    __has_builtin(__builtin_amdgcn_s_wait_tensorcnt)
#define HAVE_TDM 1
#else
#define HAVE_TDM 0
#endif
#else
#define HAVE_ASYNC_LDS 0
#define HAVE_TDM 0
#endif

// --- WMMA bf16 16x16x32, f32 accumulate -----------------------------------
__device__ __forceinline__ v8f wmma_bf16(v16bf a, v16bf b, v8f c) {
  return __builtin_amdgcn_wmma_f32_16x16x32_bf16(false, a, false, b, (short)0, c,
                                                 false, false);
}

// A fragment: 16x32 (MxK) bf16, row-major, leading dim ld (elements).
__device__ __forceinline__ v16bf frag_a(const bf16_t* __restrict__ p, int ld) {
  const int lane = threadIdx.x & 31;
  const int m = lane & 15, hh = lane >> 4;
  return *(const v16bf*)(p + m * ld + hh * 16);
}

// B fragment from TRANSPOSED storage (row-major [N][K], leading dim ld):
// one contiguous 32-byte load per lane.
__device__ __forceinline__ v16bf frag_bT(const bf16_t* __restrict__ p, int ld) {
  const int lane = threadIdx.x & 31;
  const int n = lane & 15, hh = lane >> 4;
  return *(const v16bf*)(p + n * ld + hh * 16);
}
// C/D: lane l: n = l&15; VGPR r -> m = r + 8*(l>>4)

// ---- 16-byte chunk copy: async-to-LDS when available, else VGPR bounce ----
__device__ __forceinline__ void copy16(bf16_t* lds, const bf16_t* g) {
#if HAVE_ASYNC_LDS
  __builtin_amdgcn_global_load_async_to_lds_b128(
      (__attribute__((address_space(1))) i32x4*)g,
      (__attribute__((address_space(3))) i32x4*)lds, 0, 0);
#else
  *(uint4*)lds = *(const uint4*)g;
#endif
}

__device__ __forceinline__ void stage16(bf16_t* lds, const bf16_t* g, int nelem) {
  for (int i = threadIdx.x; i < (nelem >> 3); i += 256)
    copy16(lds + i * 8, g + i * 8);
}

__device__ __forceinline__ void stage_fence() {
#if HAVE_ASYNC_LDS
  __builtin_amdgcn_s_wait_asynccnt(0);
#endif
}

// ---- TDM staging of one contiguous 32KB block (W2^T) into LDS -------------
__device__ __forceinline__ void stage_tdm_32k(bf16_t* lds, const bf16_t* g) {
#if HAVE_TDM
  if (threadIdx.x == 0) {
    const unsigned long long ga = (unsigned long long)g;
    const unsigned lo =
        (unsigned)(unsigned long)(__attribute__((address_space(3))) bf16_t*)lds;
    u32x4 g0;
    g0[0] = 1u;                                   // count=1, user, no gather
    g0[1] = lo;                                   // lds_addr
    g0[2] = (unsigned)ga;                         // global_addr[31:0]
    g0[3] = (unsigned)(ga >> 32) | (2u << 30);    // global_addr[56:32] | type=2
    i32x8 g1;
    g1[0] = (int)(3u << 16);                      // data_size = 8B
    g1[1] = (int)(4096u << 16);                   // tensor_dim0 = 4096 (8B units)
    g1[2] = (int)(1u << 16);                      // tensor_dim1 = 1
    g1[3] = (int)(4096u << 16);                   // tile_dim0 = 4096
    g1[4] = 1;                                    // tile_dim1 = 1
    g1[5] = 4096;                                 // tensor_dim0_stride = 4096
    g1[6] = 0; g1[7] = 0;
    i32x4 z4 = {0, 0, 0, 0};
#if __clang_major__ >= 23
    i32x8 z8 = {0, 0, 0, 0, 0, 0, 0, 0};
    __builtin_amdgcn_tensor_load_to_lds(g0, g1, z4, z4, z8, 0);
#else
    __builtin_amdgcn_tensor_load_to_lds(g0, g1, z4, z4, 0);
#endif
  }
  __builtin_amdgcn_s_wait_tensorcnt(0);
#else
  stage16(lds, g, CH * CH);
#endif
}

// ---------------------------------------------------------------------------
__global__ __launch_bounds__(256) void k_cvt(const float* __restrict__ src,
                                             bf16_t* __restrict__ dst, int n) {
  int i = blockIdx.x * blockDim.x + threadIdx.x;
  if (i < n) dst[i] = (bf16_t)src[i];
}

// src[rows][cols] f32 -> dst[cols][rows] bf16
__global__ __launch_bounds__(256) void k_cvt_t(const float* __restrict__ src,
                                               bf16_t* __restrict__ dst,
                                               int rows, int cols) {
  int i = blockIdx.x * blockDim.x + threadIdx.x;
  if (i < rows * cols) {
    int r = i / cols, c = i - r * cols;
    dst[c * rows + r] = (bf16_t)src[i];
  }
}

// W1 [2048=j*32+c][128=h] f32 -> W1T [j][h][c] bf16
__global__ __launch_bounds__(256) void k_cvt_w1t(const float* __restrict__ src,
                                                 bf16_t* __restrict__ dst) {
  int i = blockIdx.x * blockDim.x + threadIdx.x;
  if (i < CN * CDV * CH) {
    int row = i >> 7, h = i & 127;      // row = j*32 + c
    int j = row >> 5, c = row & 31;
    dst[(j * CH + h) * CDV + c] = (bf16_t)src[i];
  }
}

// ---------------------------------------------------------------------------
// Per-batch attention: Q/K projections + logits + softmax.  grid = B
// ---------------------------------------------------------------------------
__global__ __launch_bounds__(256)
void k_attn(const bf16_t* __restrict__ states_bf,              // [B,64,256]
            const bf16_t* __restrict__ WkT, const float* __restrict__ bk,  // [64,256]
            const bf16_t* __restrict__ WqT, const float* __restrict__ bq,  // [64,256]
            float* __restrict__ out_w,                         // [B,64,64] (d_out)
            bf16_t* __restrict__ w_bf) {                       // [B,64,64]
  __shared__ __align__(32) bf16_t s_st[CN * CS];   // 32 KB
  __shared__ __align__(32) bf16_t s_q[CN * CDK];   // Q row-major [i][dk]
  __shared__ __align__(32) bf16_t s_k[CN * CDK];   // K row-major [j][dk] (= B^T for QK^T)
  __shared__ float s_l[CN * CN];                   // logits f32
  const int b = blockIdx.x;
  stage16(s_st, states_bf + b * (CN * CS), CN * CS);
  stage_fence();
  __syncthreads();

  const int lane = threadIdx.x & 31;
  const int n_l = lane & 15, h_l = lane >> 4;

  // Q = states@Wq + bq ; K = states@Wk + bk   (32 tiles, 4 per wave)
  for (int t = WAVE * 4, e = t + 4; t < e; ++t) {
    const int mat = t >> 4;                 // 0 = K, 1 = Q
    const int mt = (t >> 2) & 3, nt = t & 3;
    const bf16_t* WT = mat ? WqT : WkT;
    v8f acc = {};
#pragma unroll
    for (int kk = 0; kk < CS / 32; ++kk)
      acc = wmma_bf16(frag_a(&s_st[(mt * 16) * CS + kk * 32], CS),
                      frag_bT(&WT[(nt * 16) * CS + kk * 32], CS), acc);
    const int c = nt * 16 + n_l;
    const float bias = mat ? bq[c] : bk[c];
    bf16_t* dst = mat ? s_q : s_k;
#pragma unroll
    for (int r = 0; r < 8; ++r) {
      const int m = mt * 16 + h_l * 8 + r;
      dst[m * CDK + c] = (bf16_t)(acc[r] + bias);
    }
  }
  __syncthreads();

  // logits = Q @ K^T / 8   (16 tiles, 2 per wave)
  for (int t = WAVE * 2, e = t + 2; t < e; ++t) {
    const int mt = t >> 2, nt = t & 3;
    v8f acc = {};
#pragma unroll
    for (int kk = 0; kk < 2; ++kk)
      acc = wmma_bf16(frag_a(&s_q[(mt * 16) * CDK + kk * 32], CDK),
                      frag_bT(&s_k[(nt * 16) * CDK + kk * 32], CDK), acc);
#pragma unroll
    for (int r = 0; r < 8; ++r)
      s_l[(mt * 16 + h_l * 8 + r) * CN + nt * 16 + n_l] = acc[r] * 0.125f;
  }
  __syncthreads();

  // row softmax (f32), one thread per row
  if (threadIdx.x < CN) {
    const int i = threadIdx.x;
    float mx = -3.402823e38f;
    for (int j = 0; j < CN; ++j) mx = fmaxf(mx, s_l[i * CN + j]);
    float sum = 0.f;
    for (int j = 0; j < CN; ++j) {
      float e = __expf(s_l[i * CN + j] - mx);
      s_l[i * CN + j] = e; sum += e;
    }
    const float inv = 1.f / sum;
    for (int j = 0; j < CN; ++j) {
      const float w = s_l[i * CN + j] * inv;
      out_w[(b * CN + i) * CN + j] = w;
      w_bf[(b * CN + i) * CN + j]  = (bf16_t)w;
    }
  }
}

// ---------------------------------------------------------------------------
// av_act / av_pol = leaky_relu([states|ext] @ Wav + bav).  grid = B
// ---------------------------------------------------------------------------
__global__ __launch_bounds__(256)
void k_av(const bf16_t* __restrict__ states_bf, const bf16_t* __restrict__ act_bf,
          const bf16_t* __restrict__ pol_bf,
          const bf16_t* __restrict__ WavT,      // [32][320] transposed
          const float* __restrict__ bav,
          bf16_t* __restrict__ av_act, bf16_t* __restrict__ av_pol) {  // [B,64,32]
  __shared__ __align__(32) bf16_t s_st[CN * CS];
  __shared__ __align__(32) bf16_t s_ex[2][CN * CA];
  const int b = blockIdx.x;
  stage16(s_st, states_bf + b * (CN * CS), CN * CS);
  stage16(s_ex[0], act_bf + b * (CN * CA), CN * CA);
  stage16(s_ex[1], pol_bf + b * (CN * CA), CN * CA);
  stage_fence();
  __syncthreads();

  const int lane = threadIdx.x & 31;
  const int n_l = lane & 15, h_l = lane >> 4;
  for (int t = WAVE * 2, e = t + 2; t < e; ++t) {    // 16 tiles: {var,mt,nt}
    const int var = t >> 3;                 // 0 act, 1 pol
    const int mt = (t >> 1) & 3, nt = t & 1;
    v8f acc = {};
#pragma unroll
    for (int kk = 0; kk < 10; ++kk) {
      v16bf a = (kk < 8)
        ? frag_a(&s_st[(mt * 16) * CS + kk * 32], CS)
        : frag_a(&s_ex[var][(mt * 16) * CA + (kk - 8) * 32], CA);
      acc = wmma_bf16(a, frag_bT(&WavT[(nt * 16) * (CS + CA) + kk * 32], CS + CA), acc);
    }
    const int c = nt * 16 + n_l;
    const float bias = bav[c];
    bf16_t* dst = (var ? av_pol : av_act) + b * (CN * CDV);
#pragma unroll
    for (int r = 0; r < 8; ++r) {
      float v = acc[r] + bias;
      v = (v > 0.f) ? v : 0.01f * v;                  // leaky_relu
      dst[(mt * 16 + h_l * 8 + r) * CDV + c] = (bf16_t)v;
    }
  }
}

// ---------------------------------------------------------------------------
// proj_base[b,j,:] = av_act[b,j,:] @ W1r[j]; delta = proj_pol - proj_base.
// grid = N (block per j): GEMM M=128(b), K=32, N=128.  projT stored [b][h][j].
// ---------------------------------------------------------------------------
__global__ __launch_bounds__(256)
void k_proj(const bf16_t* __restrict__ av_act, const bf16_t* __restrict__ av_pol,
            const bf16_t* __restrict__ W1T,     // [j][128 h][32 c]
            bf16_t* __restrict__ projT,         // [B][128 h][64 j]
            float* __restrict__ delta) {        // [B][64 j][128 h]
  __shared__ __align__(32) bf16_t s_a[CB * CDV];
  __shared__ __align__(32) bf16_t s_p[CB * CDV];
  __shared__ __align__(32) bf16_t s_b[CH * CDV];   // B^T: [h][c]
  const int j = blockIdx.x;
  for (int i = threadIdx.x; i < (CB * CDV) >> 3; i += 256) {   // 16B chunks
    const int bb = i >> 2, c8 = i & 3;                         // row bb, 8-elem chunk
    copy16(s_a + i * 8, av_act + (bb * CN + j) * CDV + c8 * 8);
    copy16(s_p + i * 8, av_pol + (bb * CN + j) * CDV + c8 * 8);
  }
  stage16(s_b, &W1T[j * CH * CDV], CH * CDV);
  stage_fence();
  __syncthreads();

  const int lane = threadIdx.x & 31;
  const int n_l = lane & 15, h_l = lane >> 4;
  for (int t = WAVE * 8, e = t + 8; t < e; ++t) {   // 64 tile-pairs
    const int mt = t >> 3, nt = t & 7;
    const v16bf bfr = frag_bT(&s_b[(nt * 16) * CDV], CDV);   // single K-step (K=32)
    v8f z = {};
    v8f accA = wmma_bf16(frag_a(&s_a[(mt * 16) * CDV], CDV), bfr, z);
    v8f accP = wmma_bf16(frag_a(&s_p[(mt * 16) * CDV], CDV), bfr, z);
    const int hcol = nt * 16 + n_l;
#pragma unroll
    for (int r = 0; r < 8; ++r) {
      const int bb = mt * 16 + h_l * 8 + r;
      const float pb = accA[r];
      projT[(bb * CH + hcol) * CN + j] = (bf16_t)pb;          // transposed for xbase
      delta[(bb * CN + j) * CH + hcol] = accP[r] - pb;
    }
  }
}

// ---------------------------------------------------------------------------
// x_base[b] = weights[b] @ proj_base[b]   ([64,64]@[64,128]).  grid = B
// ---------------------------------------------------------------------------
__global__ __launch_bounds__(256)
void k_xbase(const bf16_t* __restrict__ w_bf,     // [B,64,64]
             const bf16_t* __restrict__ projT,    // [B][128 h][64 j]  (= B^T)
             float* __restrict__ x_base) {        // [B,64,128]
  __shared__ __align__(32) bf16_t s_w[CN * CN];
  __shared__ __align__(32) bf16_t s_pT[CH * CN];
  const int b = blockIdx.x;
  stage16(s_w, w_bf + b * CN * CN, CN * CN);
  stage16(s_pT, projT + b * CH * CN, CH * CN);
  stage_fence();
  __syncthreads();

  const int lane = threadIdx.x & 31;
  const int n_l = lane & 15, h_l = lane >> 4;
  for (int t = WAVE * 4, e = t + 4; t < e; ++t) {   // 32 tiles
    const int mt = t >> 3, nt = t & 7;
    v8f acc = {};
#pragma unroll
    for (int kk = 0; kk < 2; ++kk)
      acc = wmma_bf16(frag_a(&s_w[(mt * 16) * CN + kk * 32], CN),
                      frag_bT(&s_pT[(nt * 16) * CN + kk * 32], CN), acc);
#pragma unroll
    for (int r = 0; r < 8; ++r)
      x_base[(b * CN + mt * 16 + h_l * 8 + r) * CH + nt * 16 + n_l] = acc[r];
  }
}

// ---------------------------------------------------------------------------
// Head: x = relu(x_base + w[:,q] (x) delta[q] + b1); y = relu(x@W2+b2); V = y@W3+b3.
// grid = (q=N, b=B), block = 256
// ---------------------------------------------------------------------------
__global__ __launch_bounds__(256)
void k_head(const float* __restrict__ x_base,   // [B,64,128]
            const float* __restrict__ out_w,    // [B,64,64] (f32 weights in d_out)
            const float* __restrict__ delta,    // [B,64,128]
            const float* __restrict__ b1,
            const bf16_t* __restrict__ W2T,     // [128 out][128 in] transposed
            const float* __restrict__ b2,
            const float* __restrict__ W3,       // [128]
            const float* __restrict__ b3,
            float* __restrict__ out_V) {        // [B,64,64] : V[b,p,q]
  __shared__ __align__(32) bf16_t s_x[CN * CH];    // 16 KB
  __shared__ __align__(32) bf16_t s_w2T[CH * CH];  // 32 KB
  __shared__ float s_wc[CN];
  __shared__ float s_dl[CH];
  __shared__ float s_v[CN];
  const int q = blockIdx.x, b = blockIdx.y;

  stage_tdm_32k(s_w2T, W2T);                       // TDM (or async/copy fallback)

  if (threadIdx.x < CN) {
    s_wc[threadIdx.x] = out_w[(b * CN + threadIdx.x) * CN + q];
    s_v[threadIdx.x]  = b3[0];
  } else if (threadIdx.x < CN + CH) {
    s_dl[threadIdx.x - CN] = delta[(b * CN + q) * CH + (threadIdx.x - CN)];
  }
  stage_fence();
  __syncthreads();

  for (int i = threadIdx.x; i < CN * CH; i += 256) {
    const int p = i >> 7, h = i & 127;
    float v = x_base[(b * CN + p) * CH + h] + s_wc[p] * s_dl[h] + b1[h];
    s_x[i] = (bf16_t)(v > 0.f ? v : 0.f);       // relu
  }
  __syncthreads();

  const int lane = threadIdx.x & 31;
  const int n_l = lane & 15, h_l = lane >> 4;
  for (int t = WAVE * 4, e = t + 4; t < e; ++t) {   // 32 tiles, K=128 (4 steps)
    const int mt = t >> 3, nt = t & 7;
    v8f acc = {};
#pragma unroll
    for (int kk = 0; kk < 4; ++kk)
      acc = wmma_bf16(frag_a(&s_x[(mt * 16) * CH + kk * 32], CH),
                      frag_bT(&s_w2T[(nt * 16) * CH + kk * 32], CH), acc);
    const int hcol = nt * 16 + n_l;
    const float bb2 = b2[hcol], w3 = W3[hcol];
#pragma unroll
    for (int r = 0; r < 8; ++r) {
      float y = acc[r] + bb2;
      y = y > 0.f ? y : 0.f;                      // relu, then fuse fc3 dot
      atomicAdd(&s_v[mt * 16 + h_l * 8 + r], y * w3);
    }
  }
  __syncthreads();
  if (threadIdx.x < CN)
    out_V[(b * CN + threadIdx.x) * CN + q] = s_v[threadIdx.x];
}

// ---------------------------------------------------------------------------
extern "C" void kernel_launch(void* const* d_in, const int* in_sizes, int n_in,
                              void* d_out, int out_size, void* d_ws, size_t ws_size,
                              hipStream_t stream) {
  (void)in_sizes; (void)n_in; (void)out_size; (void)ws_size;
  const float* states   = (const float*)d_in[0];
  const float* policies = (const float*)d_in[1];
  const float* actions  = (const float*)d_in[2];
  const float* Wk  = (const float*)d_in[3];  const float* bk  = (const float*)d_in[4];
  const float* Wq  = (const float*)d_in[5];  const float* bq  = (const float*)d_in[6];
  const float* Wav = (const float*)d_in[7];  const float* bav = (const float*)d_in[8];
  const float* W1  = (const float*)d_in[9];  const float* b1  = (const float*)d_in[10];
  const float* W2  = (const float*)d_in[11]; const float* b2  = (const float*)d_in[12];
  const float* W3  = (const float*)d_in[13]; const float* b3  = (const float*)d_in[14];

  float* out_V = (float*)d_out;                     // V [B,N,N,1] flat
  float* out_w = (float*)d_out + CB * CN * CN;      // weights [B,N,N]

  char* wsb = (char*)d_ws;
  size_t off = 0;
  auto carve = [&](size_t bytes) -> void* {
    void* p = wsb + off;
    off += (bytes + 255) & ~(size_t)255;
    return p;
  };
  bf16_t* states_bf = (bf16_t*)carve((size_t)CB * CN * CS * 2);
  bf16_t* act_bf    = (bf16_t*)carve((size_t)CB * CN * CA * 2);
  bf16_t* pol_bf    = (bf16_t*)carve((size_t)CB * CN * CA * 2);
  bf16_t* WkT_bf    = (bf16_t*)carve((size_t)CDK * CS * 2);
  bf16_t* WqT_bf    = (bf16_t*)carve((size_t)CDK * CS * 2);
  bf16_t* WavT_bf   = (bf16_t*)carve((size_t)CDV * (CS + CA) * 2);
  bf16_t* W1T_bf    = (bf16_t*)carve((size_t)CN * CH * CDV * 2);
  bf16_t* W2T_bf    = (bf16_t*)carve((size_t)CH * CH * 2);
  bf16_t* w_bf      = (bf16_t*)carve((size_t)CB * CN * CN * 2);
  bf16_t* av_act    = (bf16_t*)carve((size_t)CB * CN * CDV * 2);
  bf16_t* av_pol    = (bf16_t*)carve((size_t)CB * CN * CDV * 2);
  bf16_t* projT_bf  = (bf16_t*)carve((size_t)CB * CH * CN * 2);
  float*  delta     = (float*)carve((size_t)CB * CN * CH * 4);
  float*  x_base    = (float*)carve((size_t)CB * CN * CH * 4);

  auto cvt = [&](const float* s, bf16_t* d, int n) {
    k_cvt<<<(n + 255) / 256, 256, 0, stream>>>(s, d, n);
  };
  auto cvtT = [&](const float* s, bf16_t* d, int rows, int cols) {
    int n = rows * cols;
    k_cvt_t<<<(n + 255) / 256, 256, 0, stream>>>(s, d, rows, cols);
  };
  cvt(states,   states_bf, CB * CN * CS);
  cvt(actions,  act_bf,    CB * CN * CA);
  cvt(policies, pol_bf,    CB * CN * CA);
  cvtT(Wk,  WkT_bf,  CS, CDK);            // -> [64][256]
  cvtT(Wq,  WqT_bf,  CS, CDK);            // -> [64][256]
  cvtT(Wav, WavT_bf, CS + CA, CDV);       // -> [32][320]
  cvtT(W2,  W2T_bf,  CH, CH);             // -> [128][128]
  k_cvt_w1t<<<(CN * CDV * CH + 255) / 256, 256, 0, stream>>>(W1, W1T_bf);

  k_attn <<<CB, 256, 0, stream>>>(states_bf, WkT_bf, bk, WqT_bf, bq, out_w, w_bf);
  k_av   <<<CB, 256, 0, stream>>>(states_bf, act_bf, pol_bf, WavT_bf, bav, av_act, av_pol);
  k_proj <<<CN, 256, 0, stream>>>(av_act, av_pol, W1T_bf, projT_bf, delta);
  k_xbase<<<CB, 256, 0, stream>>>(w_bf, projT_bf, x_base);
  k_head <<<dim3(CN, CB), 256, 0, stream>>>(x_base, out_w, delta, b1, W2T_bf, b2, W3, b3, out_V);
}
